// LiteMLA_62302795595840
// MI455X (gfx1250) — matrix operand order repaired and checked by reading, not compiled, not run on hardware
//
#include <hip/hip_runtime.h>
#include <hip/hip_bf16.h>

typedef __attribute__((ext_vector_type(16))) __bf16 v16bf;
typedef __attribute__((ext_vector_type(8)))  float  v8f;

#define L_SP   9216      // 96*96
#define C3     768
#define CIN    256
#define COUT   256
#define CMID   512
#define NHEADS 64
#define BK     32

// ---------------------------------------------------------------------------
// K1: qkv = ReLU(BN(W[768,256] @ x[256,9216])) per batch, bf16 WMMA, f32 acc.
// Block = 256 threads (8 waves), block tile 128(M) x 64(N).
// x B-tile (32k x 64n fp32, 8KB) staged to LDS with global_load_async_to_lds
// (ASYNCcnt double-buffered); each wave: 1 A fragment -> 4 WMMAs per k-step.
// grid (144, 6, 8)
// ---------------------------------------------------------------------------
__global__ __launch_bounds__(256) void qkv_gemm_kernel(
    const float* __restrict__ x, const float* __restrict__ w,
    const float* __restrict__ gg, const float* __restrict__ bb,
    const float* __restrict__ mm, const float* __restrict__ vv,
    float* __restrict__ qkv_out)
{
    __shared__ float btile[2][BK * 64];   // 2 x 8KB double buffer

    const int t    = threadIdx.x;
    const int lane = t & 31;
    const int wave = t >> 5;
    const int hi   = lane >> 4;          // 0: lanes 0-15, 1: lanes 16-31
    const int ln   = lane & 15;
    const int l0   = blockIdx.x * 64;               // N base (spatial)
    const int m0   = blockIdx.y * 128 + wave * 16;  // M base (out channel)
    const int b    = blockIdx.z;

    const float* xb = x + (size_t)b * CIN * L_SP + l0;

    // 512 16-byte chunks per 32x64 tile; 2 per thread.
    // chunk i: row r = i>>4 (k), col group c = i&15 (4 floats)
    const unsigned r0 = (unsigned)(t >> 4),        c0 = (unsigned)(t & 15);
    const unsigned r1 = (unsigned)((t + 256) >> 4);
    const unsigned vbyte0 = (r0 * L_SP + c0 * 4) * 4u;   // global byte offset
    const unsigned vbyte1 = (r1 * L_SP + c0 * 4) * 4u;
    const unsigned lbyte0 = (r0 * 64 + c0 * 4) * 4u;     // LDS byte offset in tile
    const unsigned lbyte1 = (r1 * 64 + c0 * 4) * 4u;

    auto issue = [&](int kb, int buf) {
        const float* sp = xb + (size_t)kb * L_SP;
        const unsigned lbase = (unsigned)(size_t)(&btile[buf][0]);
        const unsigned la = lbase + lbyte0;
        const unsigned lb = lbase + lbyte1;
        asm volatile(
            "global_load_async_to_lds_b128 %0, %1, %2\n\t"
            "global_load_async_to_lds_b128 %3, %4, %2"
            :: "v"(la), "v"(vbyte0), "s"(sp), "v"(lb), "v"(vbyte1)
            : "memory");
    };

    issue(0, 0);
    v8f acc[4] = {};
    for (int kb = 0; kb < CIN; kb += BK) {
        const int  buf  = (kb >> 5) & 1;
        const bool more = (kb + BK) < CIN;
        if (more) {
            issue(kb + BK, buf ^ 1);
            // async loads retire in order: <=2 outstanding => current tile landed
            asm volatile("s_wait_asynccnt 0x2" ::: "memory");
        } else {
            asm volatile("s_wait_asynccnt 0x0" ::: "memory");
        }
        __syncthreads();

        v16bf a;
#pragma unroll
        for (int e = 0; e < 16; ++e) {
            const int ka = kb + e + (hi << 4); // A layout
            a[e] = (__bf16)w[(m0 + ln) * CIN + ka];
        }
        const float* bt = &btile[buf][0];
#pragma unroll
        for (int ns = 0; ns < 4; ++ns) {
            v16bf bfr;
#pragma unroll
            for (int e = 0; e < 16; ++e) {
                const int kk = e + (hi << 4);                          // B layout
                bfr[e] = (__bf16)bt[kk * 64 + ns * 16 + ln];
            }
            acc[ns] = __builtin_amdgcn_wmma_f32_16x16x32_bf16(
                false, a, false, bfr, (short)0, acc[ns], false, false);
        }
        __syncthreads();   // all reads of `buf` done before it is refilled
    }

    float sc[8], sh[8];
#pragma unroll
    for (int r = 0; r < 8; ++r) {
        const int m = m0 + r + (hi << 3);
        sc[r] = gg[m] * rsqrtf(vv[m] + 1e-5f);
        sh[r] = bb[m] - mm[m] * sc[r];
    }
#pragma unroll
    for (int ns = 0; ns < 4; ++ns) {
#pragma unroll
        for (int r = 0; r < 8; ++r) {
            const int m = m0 + r + (hi << 3);
            const float val = fmaxf(fmaf(acc[ns][r], sc[r], sh[r]), 0.0f);
            qkv_out[((size_t)b * C3 + m) * L_SP + l0 + ns * 16 + ln] = val;
        }
    }
}

// ---------------------------------------------------------------------------
// K2: y = grouped1x1(dw5x5(qkv)); one 8-channel group per block,
// LDS tile 8ch x (8+4) rows x (96+4) cols, grid (12, 96, 8), 256 threads
// ---------------------------------------------------------------------------
#define TH 8
__global__ __launch_bounds__(256) void dwpw_kernel(
    const float* __restrict__ qkv, const float* __restrict__ dw_w,
    const float* __restrict__ dw_b, const float* __restrict__ pw_w,
    const float* __restrict__ pw_b, float* __restrict__ y)
{
    __shared__ float tile[8][TH + 4][100];
    __shared__ float wdw[8][25];
    __shared__ float wpw[8][8];
    const int g = blockIdx.y;          // group 0..95
    const int b = blockIdx.z;
    const int row0 = blockIdx.x * TH;
    const int t = threadIdx.x;

    for (int i = t; i < 200; i += 256)
        wdw[i / 25][i % 25] = dw_w[(g * 8 + i / 25) * 25 + (i % 25)];
    if (t < 64) wpw[t >> 3][t & 7] = pw_w[(g * 8 + (t >> 3)) * 8 + (t & 7)];

    const size_t base = ((size_t)b * C3 + g * 8) * L_SP;
    for (int i = t; i < 8 * (TH + 4) * 100; i += 256) {
        const int ch = i / ((TH + 4) * 100);
        const int rem = i % ((TH + 4) * 100);
        const int rr = rem / 100, cc = rem % 100;
        const int gr = row0 + rr - 2, gc = cc - 2;
        float v = 0.0f;
        if (gr >= 0 && gr < 96 && gc >= 0 && gc < 96)
            v = qkv[base + (size_t)ch * L_SP + gr * 96 + gc];
        tile[ch][rr][cc] = v;
    }
    __syncthreads();

    for (int p = t; p < TH * 96; p += 256) {
        const int rr = p / 96, cc = p % 96;
        float z[8];
#pragma unroll
        for (int ch = 0; ch < 8; ++ch) {
            float s = dw_b[g * 8 + ch];
#pragma unroll
            for (int ky = 0; ky < 5; ++ky)
#pragma unroll
                for (int kx = 0; kx < 5; ++kx)
                    s = fmaf(tile[ch][rr + ky][cc + kx], wdw[ch][ky * 5 + kx], s);
            z[ch] = s;
        }
        const int pix = (row0 + rr) * 96 + cc;
#pragma unroll
        for (int co = 0; co < 8; ++co) {
            float s = pw_b[g * 8 + co];
#pragma unroll
            for (int ci = 0; ci < 8; ++ci) s = fmaf(wpw[co][ci], z[ci], s);
            y[base + (size_t)co * L_SP + pix] = s;
        }
    }
}

// ---------------------------------------------------------------------------
// K3: kv[b,h,d,e] = sum_l relu(k[d,l]) * v[e,l]  (v row 8 == 1)
// one block per (head, batch); deterministic shuffle+LDS tree reduce
// ---------------------------------------------------------------------------
__global__ __launch_bounds__(256) void kv_reduce_kernel(
    const float* __restrict__ qkv, const float* __restrict__ y,
    float* __restrict__ kv)
{
    const int h = blockIdx.x;   // 0..63
    const int b = blockIdx.y;
    const int t = threadIdx.x;
    const float* src = (h < 32) ? qkv : y;
    const size_t base = ((size_t)b * C3 + (h & 31) * 24) * L_SP;

    float acc[8][9];
#pragma unroll
    for (int d = 0; d < 8; ++d)
#pragma unroll
        for (int e = 0; e < 9; ++e) acc[d][e] = 0.0f;

    for (int l = t; l < L_SP; l += 256) {
        float kk[8], vvv[9];
#pragma unroll
        for (int d = 0; d < 8; ++d)
            kk[d] = fmaxf(src[base + (size_t)(8 + d) * L_SP + l], 0.0f);
#pragma unroll
        for (int e = 0; e < 8; ++e)
            vvv[e] = src[base + (size_t)(16 + e) * L_SP + l];
        vvv[8] = 1.0f;
#pragma unroll
        for (int d = 0; d < 8; ++d)
#pragma unroll
            for (int e = 0; e < 9; ++e) acc[d][e] = fmaf(kk[d], vvv[e], acc[d][e]);
    }

    __shared__ float red[8][72];
#pragma unroll
    for (int i = 0; i < 72; ++i) {
        float v = acc[i / 9][i % 9];
        for (int off = 16; off > 0; off >>= 1) v += __shfl_xor(v, off, 32);
        if ((t & 31) == 0) red[t >> 5][i] = v;
    }
    __syncthreads();
    if (t < 72) {
        float s = 0.0f;
        for (int wv = 0; wv < 8; ++wv) s += red[wv][t];
        kv[((size_t)b * NHEADS + h) * 72 + t] = s;
    }
}

// ---------------------------------------------------------------------------
// K4: attention output o (512 ch x 32 px) into LDS, then fused
// out = ReLU(BN(proj_w[256,512] @ o)) via bf16 WMMA. grid (288, 8), 256 thr.
// dynamic LDS: kv cache (64*72) + o tile (512*33, padded stride)
// ---------------------------------------------------------------------------
__global__ __launch_bounds__(256) void attn_proj_kernel(
    const float* __restrict__ qkv, const float* __restrict__ y,
    const float* __restrict__ kvg, const float* __restrict__ pw,
    const float* __restrict__ gg, const float* __restrict__ bb,
    const float* __restrict__ mm, const float* __restrict__ vv,
    float* __restrict__ out)
{
    extern __shared__ float smem[];
    float* kv_s = smem;               // 64*72
    float* o_s  = smem + NHEADS * 72; // 512*33 (stride 33: bank-conflict-free)
    const int b  = blockIdx.y;
    const int l0 = blockIdx.x * 32;
    const int t  = threadIdx.x;

    for (int i = t; i < NHEADS * 72; i += 256)
        kv_s[i] = kvg[(size_t)b * NHEADS * 72 + i];
    __syncthreads();

    // ---- attention: o[h*8+e, ll] = (sum_d kv[d,e] q[d]) / (sum_d kv[d,8] q[d] + eps)
    for (int task = t; task < NHEADS * 32; task += 256) {
        const int h = task >> 5, ll = task & 31;
        const float* src = (h < 32) ? qkv : y;
        const size_t base = ((size_t)b * C3 + (h & 31) * 24) * L_SP + l0 + ll;
        float q[8];
#pragma unroll
        for (int d = 0; d < 8; ++d)
            q[d] = fmaxf(src[base + (size_t)d * L_SP], 0.0f);
        const float* kvh = &kv_s[h * 72];
        float den = 0.0f;
#pragma unroll
        for (int d = 0; d < 8; ++d) den = fmaf(kvh[d * 9 + 8], q[d], den);
        const float inv = 1.0f / (den + 1e-12f);
#pragma unroll
        for (int e = 0; e < 8; ++e) {
            float num = 0.0f;
#pragma unroll
            for (int d = 0; d < 8; ++d) num = fmaf(kvh[d * 9 + e], q[d], num);
            o_s[(h * 8 + e) * 33 + ll] = num * inv;
        }
    }
    __syncthreads();

    // ---- proj GEMM: M=256, K=512, N=32; 16 M-tiles x 2 N-tiles over 8 waves
    const int lane = t & 31, wave = t >> 5;
    const int hi = lane >> 4, ln = lane & 15;
    for (int sub = 0; sub < 4; ++sub) {
        const int wt = wave * 4 + sub;           // 0..31
        const int m0 = (wt >> 1) * 16, n0 = (wt & 1) * 16;
        v8f acc = {};
        for (int kb = 0; kb < CMID; kb += 32) {
            v16bf a, bfr;
#pragma unroll
            for (int e = 0; e < 16; ++e) {
                const int ka = kb + e + (hi << 4);
                const int kbv = kb + e + (hi << 4);
                a[e]   = (__bf16)pw[(m0 + ln) * CMID + ka];
                bfr[e] = (__bf16)o_s[kbv * 33 + n0 + ln];
            }
            acc = __builtin_amdgcn_wmma_f32_16x16x32_bf16(
                false, a, false, bfr, (short)0, acc, false, false);
        }
#pragma unroll
        for (int r = 0; r < 8; ++r) {
            const int m = m0 + r + (hi << 3);
            const float sc = gg[m] * rsqrtf(vv[m] + 1e-5f);
            const float sh = bb[m] - mm[m] * sc;
            const float val = fmaxf(fmaf(acc[r], sc, sh), 0.0f);
            out[((size_t)b * COUT + m) * L_SP + l0 + n0 + ln] = val;
        }
    }
}

// ---------------------------------------------------------------------------
extern "C" void kernel_launch(void* const* d_in, const int* in_sizes, int n_in,
                              void* d_out, int out_size, void* d_ws, size_t ws_size,
                              hipStream_t stream)
{
    const float* x      = (const float*)d_in[0];
    const float* qkv_w  = (const float*)d_in[1];
    const float* qkv_g  = (const float*)d_in[2];
    const float* qkv_b  = (const float*)d_in[3];
    const float* qkv_m  = (const float*)d_in[4];
    const float* qkv_v  = (const float*)d_in[5];
    const float* dw_w   = (const float*)d_in[6];
    const float* dw_b   = (const float*)d_in[7];
    const float* pw_w   = (const float*)d_in[8];
    const float* pw_b   = (const float*)d_in[9];
    const float* proj_w = (const float*)d_in[10];
    const float* proj_g = (const float*)d_in[11];
    const float* proj_b = (const float*)d_in[12];
    const float* proj_m = (const float*)d_in[13];
    const float* proj_v = (const float*)d_in[14];

    float* ws  = (float*)d_ws;
    const size_t TENSOR = (size_t)8 * C3 * L_SP;  // 56,623,104 floats
    float* qkv = ws;                  // 226.5 MB
    float* yb  = ws + TENSOR;         // 226.5 MB
    float* kv  = ws + 2 * TENSOR;     // 147 KB

    qkv_gemm_kernel<<<dim3(L_SP / 64, C3 / 128, 8), 256, 0, stream>>>(
        x, qkv_w, qkv_g, qkv_b, qkv_m, qkv_v, qkv);

    dwpw_kernel<<<dim3(96 / TH, 96, 8), 256, 0, stream>>>(
        qkv, dw_w, dw_b, pw_w, pw_b, yb);

    kv_reduce_kernel<<<dim3(NHEADS, 8), 256, 0, stream>>>(qkv, yb, kv);

    const size_t smem_bytes = (size_t)(NHEADS * 72 + CMID * 33) * sizeof(float); // 86,016 B
    attn_proj_kernel<<<dim3(L_SP / 32, 8), 256, smem_bytes, stream>>>(
        qkv, yb, kv, proj_w, proj_g, proj_b, proj_m, proj_v, (float*)d_out);
}
